// AttentionAddition_80685255623116
// MI455X (gfx1250) — matrix operand order repaired and checked by reading, not compiled, not run on hardware
//
#include <hip/hip_runtime.h>
#include <stdint.h>

// ---------------------------------------------------------------------------
// Types
// ---------------------------------------------------------------------------
typedef __bf16 bf16;
typedef __attribute__((ext_vector_type(16))) __bf16 v16bf;
typedef __attribute__((ext_vector_type(8)))  float  v8f;

struct __attribute__((aligned(8))) U2 { uint32_t x, y; };

// Problem constants
#define NROI 8192
#define DIM  1024
#define SDIM 300
#define NCE  81          // C+1 class-embedding rows
#define NK   8193        // N+1 attention keys
#define SLD  8320        // padded row length for S/P and V^T (= 65*128 = 130*64)

// GEMM tiling
#define BM 128
#define BN 128
#define BK 64            // two 16x16x32 K-slices per staged tile
#define TLD 72           // LDS tile leading dim (elements), padded
#define FLAG_ACC  1
#define FLAG_RELU 2

// 1 = stage GEMM tiles with gfx1250 GLOBAL_LOAD_ASYNC_TO_LDS_B128 (ASYNCcnt)
#ifndef USE_ASYNC
#define USE_ASYNC 1
#endif

// ---------------------------------------------------------------------------
// WMMA fragment loads from LDS (CDNA5 wave32 layouts, CDNA5 ISA)
// ---------------------------------------------------------------------------
// A 16x32 bf16: lane L -> m = L&15, h = L>>4; VGPR v: K = 8h+2v (+1);
//               VGPR 4+v: K = 16+8h+2v (+1)
__device__ __forceinline__ v16bf ld_afrag(const bf16 (*sm)[TLD], int mBase,
                                          int kOff, int lane) {
  union { uint32_t u[8]; v16bf v; } r;
  const uint16_t* row = (const uint16_t*)(sm[mBase + (lane & 15)]) + kOff;
  const int h = (lane >> 4) * 8;
#pragma unroll
  for (int i = 0; i < 4; ++i) {
    r.u[i]     = *(const uint32_t*)(row + h + 2 * i);
    r.u[4 + i] = *(const uint32_t*)(row + h + 2 * i + 16);
  }
  return r.v;
}

// B 32x16 bf16 (K x N): lane L -> n = L&15, h = L>>4; VGPR v: K = 8h+2v (+1);
//               VGPR 4+v: K = 16+8h+2v (+1)
__device__ __forceinline__ v16bf ld_bfrag(const bf16 (*sm)[TLD], int nBase,
                                          int kOff, int lane) {
  union { uint32_t u[8]; v16bf v; } r;
  const uint16_t* row = (const uint16_t*)(sm[nBase + (lane & 15)]) + kOff;
  const int h = (lane >> 4) * 8;
#pragma unroll
  for (int i = 0; i < 4; ++i) {
    r.u[i]     = *(const uint32_t*)(row + h + 2 * i);
    r.u[4 + i] = *(const uint32_t*)(row + h + 2 * i + 16);
  }
  return r.v;
}

// ---------------------------------------------------------------------------
// Tile staging: global -> LDS.  Each thread owns a 32-element (64B) row chunk.
// Async path: 4x GLOBAL_LOAD_ASYNC_TO_LDS_B128 per matrix per thread; OOB rows
// zero-filled through the DS pipe. Commit = s_wait_asynccnt 0 + barrier.
// ---------------------------------------------------------------------------
__device__ __forceinline__ void stage_tile(const bf16* __restrict__ G, int ldg,
                                           int rows, int rowBase, int kk,
                                           bf16 (*sm)[TLD], int tid) {
  const int lr = tid >> 1;            // 0..127
  const int lc = (tid & 1) * 32;      // 0 / 32 elements
  const int gr = rowBase + lr;
#if USE_ASYNC
  if (gr < rows) {
    const uint64_t ga = (uint64_t)(uintptr_t)(G + (size_t)gr * ldg + kk + lc);
    const uint32_t la = (uint32_t)(uintptr_t)&sm[lr][lc];
    asm volatile("global_load_async_to_lds_b128 %0, %1, off"
                 :: "v"(la), "v"(ga) : "memory");
    asm volatile("global_load_async_to_lds_b128 %0, %1, off offset:16"
                 :: "v"(la), "v"(ga) : "memory");
    asm volatile("global_load_async_to_lds_b128 %0, %1, off offset:32"
                 :: "v"(la), "v"(ga) : "memory");
    asm volatile("global_load_async_to_lds_b128 %0, %1, off offset:48"
                 :: "v"(la), "v"(ga) : "memory");
  } else {
    U2 z = {0, 0};
    U2* d = (U2*)&sm[lr][lc];
#pragma unroll
    for (int i = 0; i < 8; ++i) d[i] = z;
  }
#else
  U2 v[8] = {};
  if (gr < rows) {
    const U2* p = (const U2*)(G + (size_t)gr * ldg + kk + lc);
#pragma unroll
    for (int i = 0; i < 8; ++i) v[i] = p[i];
  }
  U2* d = (U2*)&sm[lr][lc];
#pragma unroll
  for (int i = 0; i < 8; ++i) d[i] = v[i];
#endif
}

__device__ __forceinline__ void tile_commit() {
#if USE_ASYNC
  asm volatile("s_wait_asynccnt 0x0" ::: "memory");
#endif
  __syncthreads();
}

// ---------------------------------------------------------------------------
// Generic bf16 NT GEMM:  D[M,N] = alpha * A[M,K] @ B[N,K]^T (+bias) (+C) (relu)
//   - double-buffered LDS pipeline, 8 waves, each wave owns 32x64 of 128x128
//   - per K-slice: all 6 fragments loaded into distinct regs, then 8 WMMAs
//   - optional fp32 output C (read when FLAG_ACC), optional bf16 output Cb
// Requires K % 64 == 0 (all K here are 1024 or 8320).
// ---------------------------------------------------------------------------
extern "C" __global__ void __launch_bounds__(256)
gemm_bf16_nt(const bf16* __restrict__ A, int lda,
             const bf16* __restrict__ B, int ldb,
             float* __restrict__ C, int ldc,
             bf16* __restrict__ Cb, int ldcb,
             const float* __restrict__ bias,
             int M, int N, int K, float alpha, int flags) {
  __shared__ bf16 As[2][BM][TLD];
  __shared__ bf16 Bs[2][BN][TLD];

  const int tid   = threadIdx.x;
  const int lane  = tid & 31;
  const int wid   = tid >> 5;
  const int waveM = (wid & 3) * 32;   // 4 waves down M
  const int waveN = (wid >> 2) * 64;  // 2 waves across N
  const int blockM = blockIdx.y * BM;
  const int blockN = blockIdx.x * BN;

  v8f acc[2][4] = {};

  // prologue: stage tile 0
  stage_tile(A, lda, M, blockM, 0, As[0], tid);
  stage_tile(B, ldb, N, blockN, 0, Bs[0], tid);
  tile_commit();

  int buf = 0;
  for (int kk = 0; kk < K; kk += BK) {
    const bool more = (kk + BK) < K;
    if (more) {  // issue next tile into the other buffer (overlaps WMMA below)
      stage_tile(A, lda, M, blockM, kk + BK, As[buf ^ 1], tid);
      stage_tile(B, ldb, N, blockN, kk + BK, Bs[buf ^ 1], tid);
    }
#pragma unroll
    for (int ks = 0; ks < 2; ++ks) {   // two 16x16x32 K-slices per tile
      const int ko = ks * 32;
      // load all six fragments first (12 ds_load_b128 in flight), then WMMAs
      const v16bf a0 = ld_afrag(As[buf], waveM,      ko, lane);
      const v16bf a1 = ld_afrag(As[buf], waveM + 16, ko, lane);
      const v16bf b0 = ld_bfrag(Bs[buf], waveN +  0, ko, lane);
      const v16bf b1 = ld_bfrag(Bs[buf], waveN + 16, ko, lane);
      const v16bf b2 = ld_bfrag(Bs[buf], waveN + 32, ko, lane);
      const v16bf b3 = ld_bfrag(Bs[buf], waveN + 48, ko, lane);
      acc[0][0] = __builtin_amdgcn_wmma_f32_16x16x32_bf16(
          false, a0, false, b0, (short)0, acc[0][0], false, false);
      acc[1][0] = __builtin_amdgcn_wmma_f32_16x16x32_bf16(
          false, a1, false, b0, (short)0, acc[1][0], false, false);
      acc[0][1] = __builtin_amdgcn_wmma_f32_16x16x32_bf16(
          false, a0, false, b1, (short)0, acc[0][1], false, false);
      acc[1][1] = __builtin_amdgcn_wmma_f32_16x16x32_bf16(
          false, a1, false, b1, (short)0, acc[1][1], false, false);
      acc[0][2] = __builtin_amdgcn_wmma_f32_16x16x32_bf16(
          false, a0, false, b2, (short)0, acc[0][2], false, false);
      acc[1][2] = __builtin_amdgcn_wmma_f32_16x16x32_bf16(
          false, a1, false, b2, (short)0, acc[1][2], false, false);
      acc[0][3] = __builtin_amdgcn_wmma_f32_16x16x32_bf16(
          false, a0, false, b3, (short)0, acc[0][3], false, false);
      acc[1][3] = __builtin_amdgcn_wmma_f32_16x16x32_bf16(
          false, a1, false, b3, (short)0, acc[1][3], false, false);
    }
    if (more) { tile_commit(); buf ^= 1; }
  }

  // ---- epilogue: C 16x16 f32 layout: row = 8*(lane>>4)+r, col = lane&15 ----
  const int rhalf = (lane >> 4) * 8;
  const int lcol  = lane & 15;
#pragma unroll
  for (int mt = 0; mt < 2; ++mt) {
#pragma unroll
    for (int nt = 0; nt < 4; ++nt) {
      const int col = blockN + waveN + nt * 16 + lcol;
      if (col >= N) continue;
      const float bv = bias ? bias[col] : 0.0f;
#pragma unroll
      for (int r = 0; r < 8; ++r) {
        const int row = blockM + waveM + mt * 16 + rhalf + r;
        if (row >= M) continue;
        float v = acc[mt][nt][r] * alpha + bv;
        if (flags & FLAG_ACC)  v += C[(size_t)row * ldc + col];
        if (flags & FLAG_RELU) v = fmaxf(v, 0.0f);
        if (C)  C[(size_t)row * ldc + col]   = v;
        if (Cb) Cb[(size_t)row * ldcb + col] = (bf16)v;
      }
    }
  }
}

// ---------------------------------------------------------------------------
// Small helper kernels
// ---------------------------------------------------------------------------
extern "C" __global__ void f32_to_bf16(const float* __restrict__ s,
                                       bf16* __restrict__ d, int n) {
  int i = blockIdx.x * blockDim.x + threadIdx.x;
  if (i < n) d[i] = (bf16)s[i];
}

// ce[81,1024] = concat(class_embed, bg) @ w_proj^T + b_proj   (tiny, scalar)
extern "C" __global__ void ce_kernel(const float* __restrict__ cls,
                                     const float* __restrict__ bg,
                                     const float* __restrict__ wproj,
                                     const float* __restrict__ bproj,
                                     float* __restrict__ ce) {
  int idx = blockIdx.x * blockDim.x + threadIdx.x;
  if (idx >= NCE * DIM) return;
  const int c = idx / DIM, d = idx % DIM;
  const float* e = (c < NCE - 1) ? (cls + (size_t)c * SDIM) : bg;
  const float* w = wproj + (size_t)d * SDIM;
  float s = bproj[d];
  for (int k = 0; k < SDIM; ++k) s += e[k] * w[k];
  ce[idx] = s;
}

// gather sem per RoI + build bf16 inputs: sem, relu(sem), fp, relu(fp)
extern "C" __global__ void prep_kernel(const float* __restrict__ fp,
                                       const int* __restrict__ gt,
                                       const float* __restrict__ ce,
                                       bf16* __restrict__ sem_b,
                                       bf16* __restrict__ semr_b,
                                       bf16* __restrict__ fp_b,
                                       bf16* __restrict__ vis_b) {
  size_t idx = (size_t)blockIdx.x * blockDim.x + threadIdx.x;
  if (idx >= (size_t)NROI * DIM) return;
  const int i = (int)(idx / DIM), d = (int)(idx % DIM);
  const float s = ce[(size_t)gt[i] * DIM + d];
  const float f = fp[idx];
  sem_b[idx]  = (bf16)s;
  semr_b[idx] = (bf16)fmaxf(s, 0.0f);
  fp_b[idx]   = (bf16)f;
  vis_b[idx]  = (bf16)fmaxf(f, 0.0f);
}

// vv row N = 0 (zero V token); kp row N = dummy (raw, per reference)
extern "C" __global__ void fix_rows(bf16* __restrict__ vv, bf16* __restrict__ kp,
                                    const float* __restrict__ dummy) {
  int d = blockIdx.x * blockDim.x + threadIdx.x;
  if (d >= DIM) return;
  vv[(size_t)NROI * DIM + d] = (bf16)0.0f;
  kp[(size_t)NROI * DIM + d] = (bf16)dummy[d];
}

// vT[d][j] = vv[j][d], zero-padded to SLD columns; 32x32 LDS-tiled transpose
extern "C" __global__ void __launch_bounds__(256)
transpose_v(const bf16* __restrict__ vv, bf16* __restrict__ vT) {
  __shared__ bf16 t[32][33];
  const int tx = threadIdx.x & 31, ty = threadIdx.x >> 5;   // 32 x 8
  const int j0 = blockIdx.x * 32, d0 = blockIdx.y * 32;
#pragma unroll
  for (int i = 0; i < 4; ++i) {
    const int j = j0 + ty + i * 8;
    t[ty + i * 8][tx] = (j < NK) ? vv[(size_t)j * DIM + d0 + tx] : (bf16)0.0f;
  }
  __syncthreads();
#pragma unroll
  for (int i = 0; i < 4; ++i) {
    const int d = d0 + ty + i * 8;
    vT[(size_t)d * SLD + j0 + tx] = t[tx][ty + i * 8];
  }
}

// row softmax over NK logits (bf16 in place), zero the SLD padding
extern "C" __global__ void __launch_bounds__(256)
softmax_row(bf16* __restrict__ S) {
  __shared__ float red[256];
  const int row = blockIdx.x, tid = threadIdx.x;
  bf16* p = S + (size_t)row * SLD;

  float m = -3.0e38f;
  for (int j = tid; j < NK; j += 256) m = fmaxf(m, (float)p[j]);
  red[tid] = m; __syncthreads();
  for (int s = 128; s > 0; s >>= 1) {
    if (tid < s) red[tid] = fmaxf(red[tid], red[tid + s]);
    __syncthreads();
  }
  m = red[0]; __syncthreads();

  float sum = 0.0f;
  for (int j = tid; j < NK; j += 256) sum += __expf((float)p[j] - m);
  red[tid] = sum; __syncthreads();
  for (int s = 128; s > 0; s >>= 1) {
    if (tid < s) red[tid] += red[tid + s];
    __syncthreads();
  }
  const float inv = 1.0f / red[0];

  for (int j = tid; j < NK; j += 256)
    p[j] = (bf16)(__expf((float)p[j] - m) * inv);
  for (int j = NK + tid; j < SLD; j += 256) p[j] = (bf16)0.0f;
}

// a1 = out * relu(fp), a2 = relu(fp) - out   (bf16)
extern "C" __global__ void mix_kernel(const float* __restrict__ fp,
                                      const float* __restrict__ outb,
                                      bf16* __restrict__ a1,
                                      bf16* __restrict__ a2) {
  size_t idx = (size_t)blockIdx.x * blockDim.x + threadIdx.x;
  if (idx >= (size_t)NROI * DIM) return;
  const float v = fmaxf(fp[idx], 0.0f);
  const float o = outb[idx];
  a1[idx] = (bf16)(o * v);
  a2[idx] = (bf16)(v - o);
}

// out = relu(ffn_out + LayerNorm(o));  ffn_out already in `out` (d_out)
extern "C" __global__ void __launch_bounds__(256)
final_ln(const float* __restrict__ O, const float* __restrict__ g,
         const float* __restrict__ b, float* __restrict__ out) {
  __shared__ float red[256];
  const int row = blockIdx.x, tid = threadIdx.x;
  const float* o = O + (size_t)row * DIM;
  float x[4];
#pragma unroll
  for (int i = 0; i < 4; ++i) x[i] = o[tid + i * 256];

  red[tid] = x[0] + x[1] + x[2] + x[3]; __syncthreads();
  for (int s = 128; s > 0; s >>= 1) {
    if (tid < s) red[tid] += red[tid + s];
    __syncthreads();
  }
  const float mu = red[0] * (1.0f / DIM); __syncthreads();

  float v = 0.0f;
#pragma unroll
  for (int i = 0; i < 4; ++i) { const float d = x[i] - mu; v += d * d; }
  red[tid] = v; __syncthreads();
  for (int s = 128; s > 0; s >>= 1) {
    if (tid < s) red[tid] += red[tid + s];
    __syncthreads();
  }
  const float rs = rsqrtf(red[0] * (1.0f / DIM) + 1e-5f);

  float* op = out + (size_t)row * DIM;
#pragma unroll
  for (int i = 0; i < 4; ++i) {
    const int c = tid + i * 256;
    const float val = op[c] + g[c] * (x[i] - mu) * rs + b[c];
    op[c] = fmaxf(val, 0.0f);
  }
}

// ---------------------------------------------------------------------------
// Host-side orchestration
// ---------------------------------------------------------------------------
extern "C" void kernel_launch(void* const* d_in, const int* in_sizes, int n_in,
                              void* d_out, int out_size, void* d_ws, size_t ws_size,
                              hipStream_t stream) {
  (void)in_sizes; (void)n_in; (void)out_size; (void)ws_size;
  const float* fp    = (const float*)d_in[0];
  const int*   gt    = (const int*)  d_in[1];
  const float* cls   = (const float*)d_in[2];
  const float* bg    = (const float*)d_in[3];
  const float* wproj = (const float*)d_in[4];
  const float* bproj = (const float*)d_in[5];
  const float* wcomb = (const float*)d_in[6];
  const float* bcomb = (const float*)d_in[7];
  const float* wqk   = (const float*)d_in[8];
  const float* dummy = (const float*)d_in[9];
  const float* w1    = (const float*)d_in[10];
  const float* b1    = (const float*)d_in[11];
  const float* w2    = (const float*)d_in[12];
  const float* b2    = (const float*)d_in[13];
  const float* w3    = (const float*)d_in[14];
  const float* b3    = (const float*)d_in[15];
  const float* wf1   = (const float*)d_in[16];
  const float* bf1   = (const float*)d_in[17];
  const float* wf2   = (const float*)d_in[18];
  const float* bf2   = (const float*)d_in[19];
  const float* lng   = (const float*)d_in[20];
  const float* lnb   = (const float*)d_in[21];
  float* outp = (float*)d_out;

  // ---- workspace bump allocator ----
  char* ws = (char*)d_ws;
  size_t off = 0;
  auto alloc = [&](size_t bytes) -> char* {
    char* p = ws + off;
    off += (bytes + 255) & ~(size_t)255;
    return p;
  };
  const size_t ND = (size_t)NROI * DIM;

  bf16* wcomb_b = (bf16*)alloc((size_t)DIM * 2 * DIM * 2);
  bf16* wqk_b   = (bf16*)alloc((size_t)DIM * DIM * 2);
  bf16* w1_b    = (bf16*)alloc((size_t)(DIM / 2) * DIM * 2);
  bf16* w2_b    = (bf16*)alloc((size_t)(DIM / 2) * DIM * 2);
  bf16* w3_b    = (bf16*)alloc((size_t)DIM * 2 * DIM * 2);
  bf16* wf1_b   = (bf16*)alloc((size_t)DIM * DIM * 2);
  bf16* wf2_b   = (bf16*)alloc((size_t)DIM * DIM * 2);
  float* ce     = (float*)alloc((size_t)NCE * DIM * 4);
  bf16* sem_b   = (bf16*)alloc(ND * 2);
  bf16* semr_b  = (bf16*)alloc(ND * 2);
  bf16* fp_b    = (bf16*)alloc(ND * 2);
  bf16* vis_b   = (bf16*)alloc(ND * 2);
  bf16* qp_b    = (bf16*)alloc(ND * 2);
  bf16* kp_b    = (bf16*)alloc((size_t)NK * DIM * 2);
  bf16* vv_b    = (bf16*)alloc((size_t)NK * DIM * 2);
  float* outb   = (float*)alloc(ND * 4);
  float* obuf   = (float*)alloc(ND * 4);
  bf16* vT      = (bf16*)alloc((size_t)DIM * SLD * 2);
  bf16* Sbuf    = (bf16*)alloc((size_t)NROI * SLD * 2);
  // post-attention temporaries alias the (then dead) S buffer
  bf16* a1_b  = Sbuf;
  bf16* a2_b  = a1_b + ND;
  bf16* o12_b = a2_b + ND;
  bf16* o_b   = o12_b + ND;
  bf16* h_b   = o_b + ND;

  auto gemm = [&](const bf16* A, int lda, const bf16* B, int ldb,
                  float* C, int ldc, bf16* Cb, int ldcb, const float* bias,
                  int M, int N, int K, float alpha, int flags) {
    dim3 grid((N + BN - 1) / BN, (M + BM - 1) / BM);
    gemm_bf16_nt<<<grid, 256, 0, stream>>>(A, lda, B, ldb, C, ldc, Cb, ldcb,
                                           bias, M, N, K, alpha, flags);
  };
  auto conv = [&](const float* s, bf16* d, size_t n) {
    f32_to_bf16<<<(unsigned)((n + 255) / 256), 256, 0, stream>>>(s, d, (int)n);
  };

  // 1) weights -> bf16
  conv(wcomb, wcomb_b, (size_t)DIM * 2 * DIM);
  conv(wqk,   wqk_b,   (size_t)DIM * DIM);
  conv(w1,    w1_b,    (size_t)(DIM / 2) * DIM);
  conv(w2,    w2_b,    (size_t)(DIM / 2) * DIM);
  conv(w3,    w3_b,    (size_t)DIM * 2 * DIM);
  conv(wf1,   wf1_b,   (size_t)DIM * DIM);
  conv(wf2,   wf2_b,   (size_t)DIM * DIM);

  // 2) class embedding projection + per-RoI gather
  ce_kernel<<<(NCE * DIM + 255) / 256, 256, 0, stream>>>(cls, bg, wproj, bproj, ce);
  prep_kernel<<<(unsigned)(ND / 256), 256, 0, stream>>>(fp, gt, ce, sem_b, semr_b,
                                                        fp_b, vis_b);

  // 3) comb = sem @ Wc[:, :D]^T + fp @ Wc[:, D:]^T + b_comb  -> vv (bf16)
  gemm(sem_b, DIM, wcomb_b,       2 * DIM, outb, DIM, nullptr, 0, bcomb,
       NROI, DIM, DIM, 1.0f, 0);
  gemm(fp_b,  DIM, wcomb_b + DIM, 2 * DIM, outb, DIM, vv_b, DIM, nullptr,
       NROI, DIM, DIM, 1.0f, FLAG_ACC);

  // 4) qp = (vis @ Wqk^T) / sqrt(D);  kp = relu(sem) @ Wqk^T; append dummy/zero
  gemm(vis_b,  DIM, wqk_b, DIM, nullptr, 0, qp_b, DIM, nullptr,
       NROI, DIM, DIM, 1.0f / 32.0f, 0);
  gemm(semr_b, DIM, wqk_b, DIM, nullptr, 0, kp_b, DIM, nullptr,
       NROI, DIM, DIM, 1.0f, 0);
  fix_rows<<<(DIM + 255) / 256, 256, 0, stream>>>(vv_b, kp_b, dummy);

  // 5) S = qp @ kp^T  (bf16 logits, padded row stride SLD)
  gemm(qp_b, DIM, kp_b, DIM, nullptr, 0, Sbuf, SLD, nullptr,
       NROI, NK, DIM, 1.0f, 0);
  softmax_row<<<NROI, 256, 0, stream>>>(Sbuf);

  // 6) out = P @ V  (V transposed + zero-padded to SLD so K % 64 == 0)
  {
    dim3 tg(SLD / 32, DIM / 32);
    transpose_v<<<tg, 256, 0, stream>>>(vv_b, vT);
  }
  gemm(Sbuf, SLD, vT, SLD, outb, DIM, nullptr, 0, nullptr,
       NROI, DIM, SLD, 1.0f, 0);

  // 7) residual branches
  mix_kernel<<<(unsigned)(ND / 256), 256, 0, stream>>>(fp, outb, a1_b, a2_b);
  gemm(a1_b, DIM, w1_b, DIM, nullptr, 0, o12_b,           DIM, b1,
       NROI, DIM / 2, DIM, 1.0f, FLAG_RELU);
  gemm(a2_b, DIM, w2_b, DIM, nullptr, 0, o12_b + DIM / 2, DIM, b2,
       NROI, DIM / 2, DIM, 1.0f, FLAG_RELU);

  // 8) o = [o1,o2] @ W3[:, :D]^T + vis @ W3[:, D:]^T + b3
  gemm(o12_b, DIM, w3_b,       2 * DIM, obuf, DIM, nullptr, 0, b3,
       NROI, DIM, DIM, 1.0f, 0);
  gemm(vis_b, DIM, w3_b + DIM, 2 * DIM, obuf, DIM, o_b, DIM, nullptr,
       NROI, DIM, DIM, 1.0f, FLAG_ACC);

  // 9) FFN:  h = relu(o @ Wf1^T + bf1);  ffn = h @ Wf2^T + bf2 -> d_out
  gemm(o_b, DIM, wf1_b, DIM, nullptr, 0, h_b, DIM, bf1,
       NROI, DIM, DIM, 1.0f, FLAG_RELU);
  gemm(h_b, DIM, wf2_b, DIM, outp, DIM, nullptr, 0, bf2,
       NROI, DIM, DIM, 1.0f, 0);

  // 10) d_out = relu(ffn + LayerNorm(o))
  final_ln<<<NROI, 256, 0, stream>>>(obuf, lng, lnb, outp);
}
